// Head_29291676958807
// MI455X (gfx1250) — matrix operand order repaired and checked by reading, not compiled, not run on hardware
//
#include <hip/hip_runtime.h>
#include <hip/hip_bf16.h>

// ---------------------------------------------------------------------------
// Fused attention head for MI455X (gfx1250, wave32, WMMA).
//   inputs : x [8,2048,1024] f32, Wq/Wk/Wv [1024,256] f32
//   output : [8,2048,256] f32
//
// Pipeline:
//   0a. cvt_x : x fp32 -> xb bf16          0b. cvt_w : W -> Wt bf16 transposed
//   1.  qkv_gemm : Qb,Kb ([b][t][d]), Vt ([b][d][t]) via bf16 WMMA
//   2.  attn : causal flash attention; K/V blocks DOUBLE-BUFFERED in LDS via
//       GLOBAL_LOAD_ASYNC_TO_LDS_B128 + counted s_wait_asynccnt (in-order
//       completion), overlapping next-tile DMA with current-tile WMMA.
// ---------------------------------------------------------------------------

#define NEMB 1024
#define HD   256
#define BATCH 8
#define SEQ  2048
#define BT   (BATCH * SEQ)

// LDS tile geometry (padded for conflict-free ds_load_b128)
#define KT_PITCH 264   // ushorts per K-tile row (256 data + 8 pad = 528 B)
#define VT_PITCH 40    // ushorts per V-tile row ( 32 data + 8 pad =  80 B)

typedef __attribute__((ext_vector_type(16))) __bf16          v16bf;
typedef __attribute__((ext_vector_type(16))) unsigned short  v16us;
typedef __attribute__((ext_vector_type(8)))  unsigned short  v8us;
typedef __attribute__((ext_vector_type(4)))  unsigned short  v4us;
typedef __attribute__((ext_vector_type(4)))  int             v4i;
typedef __attribute__((ext_vector_type(4)))  float           v4f;
typedef __attribute__((ext_vector_type(8)))  float           v8f;

union frag16 { v16us v; v8us h[2]; };

__device__ __forceinline__ unsigned short f2bf(float f) {
  unsigned u = __builtin_bit_cast(unsigned, f);
  unsigned r = u + 0x7FFFu + ((u >> 16) & 1u);   // round-to-nearest-even
  return (unsigned short)(r >> 16);
}

__device__ __forceinline__ v8f wmma_bf16(const v16us& a, const v16us& b, v8f c) {
  return __builtin_amdgcn_wmma_f32_16x16x32_bf16(
      false, __builtin_bit_cast(v16bf, a),
      false, __builtin_bit_cast(v16bf, b),
      (short)0, c, false, false);
}

// --- async global->LDS copy of one 16B chunk (per lane), with sync fallback --
#if defined(__has_builtin) && __has_builtin(__builtin_amdgcn_global_load_async_to_lds_b128)
#define HAVE_ASYNC_LDS 1
__device__ __forceinline__ void copy16_async(const unsigned short* g, unsigned short* l) {
  __builtin_amdgcn_global_load_async_to_lds_b128(
      (v4i*)(void*)g, (v4i*)(void*)l, 0, 0);
}
// wait until at most `n` async ops remain in flight (in-order completion
// guarantees the OLDER batch has finished)
__device__ __forceinline__ void wait_async_le16() {
  asm volatile("s_wait_asynccnt 0x10" ::: "memory");
}
__device__ __forceinline__ void wait_async_le0() {
  asm volatile("s_wait_asynccnt 0x0" ::: "memory");
}
#else
#define HAVE_ASYNC_LDS 0
__device__ __forceinline__ void copy16_async(const unsigned short* g, unsigned short* l) {
  *(v8us*)l = *(const v8us*)g;
}
__device__ __forceinline__ void wait_async_le16() {}
__device__ __forceinline__ void wait_async_le0()  {}
#endif

// ---------------------------------------------------------------------------
// Kernel 0a: x fp32 -> bf16
// ---------------------------------------------------------------------------
__global__ __launch_bounds__(256) void cvt_x(const float* __restrict__ x,
                                             unsigned short* __restrict__ xb) {
  size_t i = ((size_t)blockIdx.x * 256 + threadIdx.x) * 4;
  v4f f = *(const v4f*)(x + i);
  v4us r;
  r[0] = f2bf(f[0]); r[1] = f2bf(f[1]); r[2] = f2bf(f[2]); r[3] = f2bf(f[3]);
  *(v4us*)(xb + i) = r;
}

// ---------------------------------------------------------------------------
// Kernel 0b: Wq/Wk/Wv fp32 [k][d] -> Wt bf16 [mat][d][k]  (transposed)
// ---------------------------------------------------------------------------
__global__ __launch_bounds__(256) void cvt_w(const float* __restrict__ Wq,
                                             const float* __restrict__ Wk,
                                             const float* __restrict__ Wv,
                                             unsigned short* __restrict__ wt) {
  int idx = blockIdx.x * 256 + threadIdx.x;
  int mat = idx >> 18;
  int r   = idx & ((HD * NEMB) - 1);
  int d   = r >> 10;
  int k   = r & (NEMB - 1);
  const float* W = (mat == 0) ? Wq : ((mat == 1) ? Wk : Wv);
  wt[idx] = f2bf(W[(size_t)k * HD + d]);
}

// ---------------------------------------------------------------------------
// Kernel 1: Q/K/V projection (vector bf16 loads + WMMA pipeline)
// ---------------------------------------------------------------------------
__global__ __launch_bounds__(256) void qkv_gemm(
    const unsigned short* __restrict__ xb,
    const unsigned short* __restrict__ wt,
    unsigned short* __restrict__ qkv) {
  const int w    = threadIdx.x >> 5;
  const int lane = threadIdx.x & 31;
  const int m    = lane & 15;
  const int hi   = lane >> 4;

  const int ct   = blockIdx.y * 8 + w;
  const int mat  = ct >> 4;
  const int col0 = (ct & 15) << 4;
  const int row0 = blockIdx.x << 4;

  const unsigned short* Wt   = wt + (size_t)mat * HD * NEMB;
  const unsigned short* arow = xb + (size_t)(row0 + m) * NEMB;
  const unsigned short* brow = Wt + (size_t)(col0 + m) * NEMB;
  unsigned short* outp = qkv + (size_t)mat * BT * HD;

  v8f acc = {};
  for (int k0 = 0; k0 < NEMB; k0 += 32) {
    frag16 a;
    a.h[0] = *(const v8us*)(arow + k0 + hi * 16);
    a.h[1] = *(const v8us*)(arow + k0 + hi * 16 + 8);
    v16us bm = *(const v16us*)(brow + k0 + hi * 16);
    acc = wmma_bf16(a.v, bm, acc);
  }

#pragma unroll
  for (int vg = 0; vg < 8; ++vg) {
    int r = row0 + vg + hi * 8;
    unsigned short v = f2bf(acc[vg]);
    if (mat < 2) {
      outp[(size_t)r * HD + col0 + m] = v;
    } else {
      int bb = r >> 11, t = r & (SEQ - 1);
      outp[((size_t)(bb * HD + col0 + m)) * SEQ + t] = v;
    }
  }
}

// ---------------------------------------------------------------------------
// Kernel 2: causal flash attention, double-buffered async K/V staging.
// Per iteration jb:
//   issue async copies for tile jb+1  -> buf[(jb+1)&1]   (16 ops/thread)
//   s_wait_asynccnt 16   (older batch done, new batch still in flight)
//   barrier; compute tile jb from buf[jb&1] (32 WMMAs); barrier
// ---------------------------------------------------------------------------
__global__ __launch_bounds__(128) void attn(
    const unsigned short* __restrict__ Qb,
    const unsigned short* __restrict__ Kb,
    const unsigned short* __restrict__ Vt,
    float* __restrict__ out) {
  __shared__ __align__(16) unsigned short ktile[2][32][KT_PITCH];  // ~33.8 KB
  __shared__ __align__(16) unsigned short vtile[2][HD][VT_PITCH];  // ~40.0 KB
  __shared__ __align__(16) unsigned short pbuf[4][16][40];         //  ~5.1 KB

  const int tid  = threadIdx.x;
  const int w    = tid >> 5;
  const int lane = tid & 31;
  const int m    = lane & 15;
  const int hi   = lane >> 4;
  const int b    = blockIdx.y;
  const int qbase = blockIdx.x * 64;
  const int q0    = qbase + w * 16;

  const unsigned short* Qp = Qb + (size_t)b * SEQ * HD;
  const unsigned short* Kp = Kb + (size_t)b * SEQ * HD;
  const unsigned short* Vp = Vt + (size_t)b * HD * SEQ;

  const int nkb_blk = ((qbase + 63) >> 5) + 1;     // key blocks for whole block

  // --- issue async stage of tile 0 first, then load Q while DMA runs ---
  auto stage = [&](int jb, int buf) {
    const int kb = jb * 32;
    // K: 32 rows x 512B = 1024 chunks of 16B
#pragma unroll
    for (int it = 0; it < 8; ++it) {
      int id = it * 128 + tid;
      int r  = id >> 5;
      int c8 = (id & 31) * 8;
      copy16_async(Kp + (size_t)(kb + r) * HD + c8, &ktile[buf][r][c8]);
    }
    // V: 256 rows x 64B = 1024 chunks of 16B
#pragma unroll
    for (int it = 0; it < 8; ++it) {
      int id = it * 128 + tid;
      int d  = id >> 2;
      int c8 = (id & 3) * 8;
      copy16_async(Vp + (size_t)d * SEQ + kb + c8, &vtile[buf][d][c8]);
    }
  };
  stage(0, 0);

  // Q fragments for all 8 d-chunks (registers, contiguous vector loads)
  frag16 qa[8];
  {
    const unsigned short* qrow = Qp + (size_t)(q0 + m) * HD;
#pragma unroll
    for (int c = 0; c < 8; ++c) {
      qa[c].h[0] = *(const v8us*)(qrow + c * 32 + hi * 16);
      qa[c].h[1] = *(const v8us*)(qrow + c * 32 + hi * 16 + 8);
    }
  }

  v8f o[16];
#pragma unroll
  for (int n = 0; n < 16; ++n) o[n] = (v8f){};
  float mrow[8], lrow[8];
#pragma unroll
  for (int vg = 0; vg < 8; ++vg) { mrow[vg] = -1e30f; lrow[vg] = 0.f; }

  const float scale = 0.03125f;                    // 1/sqrt(1024)

  for (int jb = 0; jb < nkb_blk; ++jb) {
    const int kb  = jb * 32;
    const int buf = jb & 1;

    // prefetch next tile into the other buffer, then wait for current tile
    if (jb + 1 < nkb_blk) {
      stage(jb + 1, buf ^ 1);
      wait_async_le16();                 // older 16 ops (tile jb) complete
    } else {
      wait_async_le0();
    }
    __syncthreads();                     // all waves' tile-jb copies visible

    if (kb <= q0 + 15) {                 // causal: this wave still has work
      // ---- scores S = Q K^T from LDS (conflict-free b128 reads) ----
      v8f s0 = {}, s1 = {};
      const unsigned short* k0p = &ktile[buf][m][hi * 16];
      const unsigned short* k1p = &ktile[buf][16 + m][hi * 16];
#pragma unroll
      for (int c = 0; c < 8; ++c) {
        v16us kf0 = *(const v16us*)(k0p + c * 32);
        v16us kf1 = *(const v16us*)(k1p + c * 32);
        s0 = wmma_bf16(qa[c].v, kf0, s0);
        s1 = wmma_bf16(qa[c].v, kf1, s1);
      }

      // ---- online softmax (row = vg + hi*8, col = m across 16 lanes) ----
      float p0v[8], p1v[8], alpha[8];
#pragma unroll
      for (int vg = 0; vg < 8; ++vg) {
        int qrow = q0 + vg + hi * 8;
        float v0 = s0[vg] * scale;
        float v1 = s1[vg] * scale;
        if (kb + m      > qrow) v0 = -1e30f;
        if (kb + 16 + m > qrow) v1 = -1e30f;
        float rm = fmaxf(v0, v1);
#pragma unroll
        for (int off = 1; off < 16; off <<= 1) rm = fmaxf(rm, __shfl_xor(rm, off));
        float mnew = fmaxf(mrow[vg], rm);
        float p0 = __expf(v0 - mnew);
        float p1 = __expf(v1 - mnew);
        float rs = p0 + p1;
#pragma unroll
        for (int off = 1; off < 16; off <<= 1) rs += __shfl_xor(rs, off);
        alpha[vg] = __expf(mrow[vg] - mnew);
        mrow[vg]  = mnew;
        lrow[vg]  = lrow[vg] * alpha[vg] + rs;
        p0v[vg] = p0;
        p1v[vg] = p1;
      }
#pragma unroll
      for (int n = 0; n < 16; ++n)
#pragma unroll
        for (int vg = 0; vg < 8; ++vg) o[n][vg] *= alpha[vg];

      // ---- relayout P: C-layout -> A-layout via per-wave LDS ----
#pragma unroll
      for (int vg = 0; vg < 8; ++vg) {
        int row = vg + hi * 8;
        pbuf[w][row][m]      = f2bf(p0v[vg]);
        pbuf[w][row][16 + m] = f2bf(p1v[vg]);
      }
      asm volatile("s_wait_dscnt 0" ::: "memory");
      frag16 pa;
      pa.h[0] = *(const v8us*)(&pbuf[w][m][hi * 16]);
      pa.h[1] = *(const v8us*)(&pbuf[w][m][hi * 16 + 8]);

      // ---- O += P V from LDS ----
#pragma unroll
      for (int n = 0; n < 16; ++n) {
        v16us vf = *(const v16us*)(&vtile[buf][n * 16 + m][hi * 16]);
        o[n] = wmma_bf16(pa.v, vf, o[n]);
      }
    }
    __syncthreads();   // compute(jb) done everywhere before buf is re-staged
  }

  // ---- normalize and store fp32 output ----
  float inv[8];
#pragma unroll
  for (int vg = 0; vg < 8; ++vg) inv[vg] = 1.0f / lrow[vg];
#pragma unroll
  for (int n = 0; n < 16; ++n)
#pragma unroll
    for (int vg = 0; vg < 8; ++vg) {
      int row = q0 + vg + hi * 8;
      out[((size_t)b * SEQ + row) * HD + n * 16 + m] = o[n][vg] * inv[vg];
    }
}

// ---------------------------------------------------------------------------
extern "C" void kernel_launch(void* const* d_in, const int* in_sizes, int n_in,
                              void* d_out, int out_size, void* d_ws, size_t ws_size,
                              hipStream_t stream) {
  const float* x  = (const float*)d_in[0];
  const float* Wq = (const float*)d_in[1];
  const float* Wk = (const float*)d_in[2];
  const float* Wv = (const float*)d_in[3];
  float* out = (float*)d_out;

  // Workspace layout (bf16 = ushort): Qb | Kb | Vt | xb | Wt
  unsigned short* ws  = (unsigned short*)d_ws;
  unsigned short* Qb  = ws;
  unsigned short* Kb  = ws + (size_t)1 * BT * HD;
  unsigned short* Vt  = ws + (size_t)2 * BT * HD;
  unsigned short* xb  = ws + (size_t)3 * BT * HD;
  unsigned short* wt  = ws + (size_t)3 * BT * HD + (size_t)BT * NEMB;

  cvt_x<<<dim3((BT * NEMB) / (256 * 4)), 256, 0, stream>>>(x, xb);
  cvt_w<<<dim3((3 * HD * NEMB) / 256),   256, 0, stream>>>(Wq, Wk, Wv, wt);
  qkv_gemm<<<dim3(BT / 16, 6), 256, 0, stream>>>(xb, wt, ws);
  attn<<<dim3(SEQ / 64, BATCH), 128, 0, stream>>>(Qb, Kb, Vt, out);
}
